// _PairwiseScoring_59571196395588
// MI455X (gfx1250) — matrix operand order repaired and checked by reading, not compiled, not run on hardware
//
#include <hip/hip_runtime.h>

typedef _Float16 v8h  __attribute__((ext_vector_type(8)));
typedef _Float16 v16h __attribute__((ext_vector_type(16)));
typedef float    v8f  __attribute__((ext_vector_type(8)));

#define B_SZ   2
#define N_SZ   1024
#define D_SZ   512
#define P_SZ   128
#define H_SZ   64

// workspace byte offsets (all 256B aligned)
#define OFF_W1F    0u          // 16 frags * 32 lanes * 16 halfs * 2B = 16384
#define OFF_ROWH   32768u      // 2048*128*2 = 524288
#define OFF_COLH   557056u     // 2048*128*2 = 524288
#define OFF_SCORES 1081344u    // 2*1024*1024*4 = 8388608

__device__ __forceinline__ v16h cat8(v8h lo, v8h hi) {
    return __builtin_shufflevector(lo, hi, 0,1,2,3,4,5,6,7,8,9,10,11,12,13,14,15);
}

// ---------------------------------------------------------------------------
// K0: repack W1 (128x64 f32, row-major) into WMMA B-fragment order, f16.
// frag f = ntile*4+kc ; per lane 16 halfs: element e -> K = kc*32 + (lane<16?0:16) + e,
// N = ntile*16 + lane%16. Stored at halfs[(f*32+lane)*16 + e].
// ---------------------------------------------------------------------------
__global__ void k0_pack_w1(const float* __restrict__ W1, _Float16* __restrict__ w1f) {
    int t = blockIdx.x * blockDim.x + threadIdx.x;
    if (t >= 16 * 32) return;
    int frag = t >> 5, lane = t & 31;
    int ntile = frag >> 2, kc = frag & 3;
    int koff = kc * 32 + ((lane & 16) ? 16 : 0);
    int n = ntile * 16 + (lane & 15);
    #pragma unroll
    for (int e = 0; e < 16; ++e)
        w1f[t * 16 + e] = (_Float16)W1[(koff + e) * H_SZ + n];
}

// ---------------------------------------------------------------------------
// K1: row = x@Wr + br ; col = x@Wc + bc, stored as f16. One block per (b,n).
// ---------------------------------------------------------------------------
__global__ void k1_proj(const float* __restrict__ x,
                        const float* __restrict__ Wr, const float* __restrict__ br,
                        const float* __restrict__ Wc, const float* __restrict__ bc,
                        _Float16* __restrict__ rowh, _Float16* __restrict__ colh) {
    __shared__ float xs[D_SZ];
    int bn = blockIdx.x;                 // 0..2047
    int p  = threadIdx.x;                // 0..127
    for (int d = threadIdx.x; d < D_SZ; d += blockDim.x)
        xs[d] = x[(size_t)bn * D_SZ + d];
    __syncthreads();
    float ar = br[p], ac = bc[p];
    #pragma unroll 4
    for (int d = 0; d < D_SZ; ++d) {
        float xv = xs[d];
        ar = fmaf(xv, Wr[d * P_SZ + p], ar);
        ac = fmaf(xv, Wc[d * P_SZ + p], ac);
    }
    rowh[(size_t)bn * P_SZ + p] = (_Float16)ar;
    colh[(size_t)bn * P_SZ + p] = (_Float16)ac;
}

// ---------------------------------------------------------------------------
// K2: main fused pair-MLP. Persistent waves; each tile = (b, i, j0..j0+15).
// GEMM: M=16 pairs, N=64 hidden (4 tiles), K=128 (4 x wmma_f32_16x16x32_f16).
// A built on the fly: pair = row[i] + col[j] (packed f16 adds).
// Epilogue: exact-erf GELU, dot with W2 via shfl_xor butterfly, +b2.
// ---------------------------------------------------------------------------
__global__ void __launch_bounds__(256, 1)
k2_main(const v8h* __restrict__ rowh, const v8h* __restrict__ colh,
        const v8h* __restrict__ w1f,
        const float* __restrict__ W2, const float* __restrict__ b1,
        const float* __restrict__ b2, float* __restrict__ scores,
        int tiles_total) {
    const int lane = threadIdx.x & 31;
    const int wave = blockIdx.x * (blockDim.x >> 5) + (threadIdx.x >> 5);
    const int nwaves = gridDim.x * (blockDim.x >> 5);
    const int l16  = lane & 15;
    const int hoff = (lane >> 4) & 1;    // A-fragment half select

    // resident W1 B-fragments: bf[ntile*4+kc]
    v16h bf[16];
    #pragma unroll
    for (int f = 0; f < 16; ++f) {
        int base = (f * 32 + lane) * 2;  // v8h units
        bf[f] = cat8(w1f[base], w1f[base + 1]);
    }
    float w2v[4], b1v[4];
    #pragma unroll
    for (int t = 0; t < 4; ++t) {
        w2v[t] = W2[t * 16 + l16];
        b1v[t] = b1[t * 16 + l16];
    }
    const float b2v = b2[0];

    for (int tile = wave; tile < tiles_total; tile += nwaves) {
        const int b   = tile >> 16;            // 65536 tiles per batch
        const int rem = tile & 0xFFFF;
        const int i   = rem >> 6;
        const int j0  = (rem & 63) << 4;
        const int rbase = ((b << 10) + i) * 16;          // v8h units
        const int cbase = ((b << 10) + j0 + l16) * 16;

        v8f acc[4];
        #pragma unroll
        for (int t = 0; t < 4; ++t) {
            float bb = b1v[t];
            acc[t] = (v8f){bb, bb, bb, bb, bb, bb, bb, bb};
        }
        #pragma unroll
        for (int kc = 0; kc < 4; ++kc) {
            v8h r0 = rowh[rbase + kc * 4 + 2 * hoff];
            v8h r1 = rowh[rbase + kc * 4 + 2 * hoff + 1];
            v8h c0 = colh[cbase + kc * 4 + 2 * hoff];
            v8h c1 = colh[cbase + kc * 4 + 2 * hoff + 1];
            v16h a = cat8(r0 + c0, r1 + c1);             // v_pk_add_f16
            #pragma unroll
            for (int t = 0; t < 4; ++t)
                acc[t] = __builtin_amdgcn_wmma_f32_16x16x32_f16(
                    false, a, false, bf[t * 4 + kc], (short)0, acc[t], false, false);
        }
        // GELU (exact erf) + fold W2
        float p[8] = {0.f, 0.f, 0.f, 0.f, 0.f, 0.f, 0.f, 0.f};
        #pragma unroll
        for (int t = 0; t < 4; ++t) {
            #pragma unroll
            for (int e = 0; e < 8; ++e) {
                float xv = acc[t][e];
                float g = 0.5f * xv * (1.0f + erff(xv * 0.70710678118654752f));
                p[e] = fmaf(g, w2v[t], p[e]);
            }
        }
        // reduce over N-lanes (16-lane halves), lanes<16 -> j'=0..7, >=16 -> j'=8..15
        #pragma unroll
        for (int e = 0; e < 8; ++e) {
            p[e] += __shfl_xor(p[e], 1, 32);
            p[e] += __shfl_xor(p[e], 2, 32);
            p[e] += __shfl_xor(p[e], 4, 32);
            p[e] += __shfl_xor(p[e], 8, 32);
            p[e] += b2v;
        }
        float* orow = scores + (((size_t)b << 10) + i) * N_SZ + j0;
        if (lane == 0) {
            *(float4*)(orow)     = make_float4(p[0], p[1], p[2], p[3]);
            *(float4*)(orow + 4) = make_float4(p[4], p[5], p[6], p[7]);
        } else if (lane == 16) {
            *(float4*)(orow + 8)  = make_float4(p[0], p[1], p[2], p[3]);
            *(float4*)(orow + 12) = make_float4(p[4], p[5], p[6], p[7]);
        }
    }
}

// ---------------------------------------------------------------------------
// K3: out[b,i,j] = 0.5*(s[b,i,j] + s[b,j,i])
// ---------------------------------------------------------------------------
__global__ void k3_sym(const float* __restrict__ s, float* __restrict__ out) {
    size_t tid = (size_t)blockIdx.x * blockDim.x + threadIdx.x;
    size_t b   = tid >> 20;
    size_t rem = tid & 0xFFFFFu;
    size_t i   = rem >> 10;
    size_t j   = rem & 1023u;
    out[tid] = 0.5f * (s[tid] + s[(b << 20) + (j << 10) + i]);
}

extern "C" void kernel_launch(void* const* d_in, const int* in_sizes, int n_in,
                              void* d_out, int out_size, void* d_ws, size_t ws_size,
                              hipStream_t stream) {
    const float* x  = (const float*)d_in[0];
    const float* Wr = (const float*)d_in[1];
    const float* br = (const float*)d_in[2];
    const float* Wc = (const float*)d_in[3];
    const float* bc = (const float*)d_in[4];
    const float* W1 = (const float*)d_in[5];
    const float* b1 = (const float*)d_in[6];
    const float* W2 = (const float*)d_in[7];
    const float* b2 = (const float*)d_in[8];

    char* ws = (char*)d_ws;
    _Float16* w1f  = (_Float16*)(ws + OFF_W1F);
    _Float16* rowh = (_Float16*)(ws + OFF_ROWH);
    _Float16* colh = (_Float16*)(ws + OFF_COLH);
    float*    scr  = (float*)(ws + OFF_SCORES);

    k0_pack_w1<<<2, 256, 0, stream>>>(W1, w1f);
    k1_proj<<<B_SZ * N_SZ, P_SZ, 0, stream>>>(x, Wr, br, Wc, bc, rowh, colh);

    const int tiles_total = B_SZ * N_SZ * (N_SZ / 16);   // 131072
    k2_main<<<2048, 256, 0, stream>>>((const v8h*)rowh, (const v8h*)colh,
                                      (const v8h*)w1f, W2, b1, b2, scr, tiles_total);

    k3_sym<<<(B_SZ * N_SZ * N_SZ) / 256, 256, 0, stream>>>(scr, (float*)d_out);
}
